// Bond2BondLayer_41532333752516
// MI455X (gfx1250) — compile-verified, matrix operand order checked by reading, not run on hardware
//
#include <hip/hip_runtime.h>
#include <hip/hip_bf16.h>

typedef __bf16 bf16;
typedef __attribute__((ext_vector_type(16))) __bf16 v16bf;
typedef __attribute__((ext_vector_type(8)))  __bf16 v8bf;
typedef __attribute__((ext_vector_type(8)))  float  v8f;

#define PI_F 3.1415926f

// ---------------------------------------------------------------- WMMA core
__device__ __forceinline__ v8f wmma_bf16(v16bf a, v16bf b, v8f c) {
  // (neg_a, A, neg_b, B, c_mod, C, reuse_a, reuse_b)
  return __builtin_amdgcn_wmma_f32_16x16x32_bf16(false, a, false, b, (short)0, c,
                                                 false, false);
}

// A-tile (16x32 bf16, row-major source). ISA layout: lanes 0-15 hold row M=lane,
// K = kBase+0..7 (VGPR0-3) and kBase+16..23 (VGPR4-7); lanes 16-31 hold the same
// row's K = kBase+8..15 / kBase+24..31. Two 16-byte loads per lane.
__device__ __forceinline__ v16bf load_a16x32(const bf16* __restrict__ A, int lda,
                                             int mBase, int kBase, int lane) {
  const int row = mBase + (lane & 15);
  const int k0  = kBase + ((lane >> 4) << 3);
  const bf16* p = A + (size_t)row * lda + k0;
  v8bf lo = *(const v8bf*)(p);
  v8bf hi = *(const v8bf*)(p + 16);
  return __builtin_shufflevector(lo, hi, 0,1,2,3,4,5,6,7,8,9,10,11,12,13,14,15);
}

// C = A(bf16 row-major MxK) * Bpacked(bf16, WMMA lane layout KxN), optional relu,
// optional bf16 output. One wave per 16-row strip; 8 N-tiles (128 cols) per chunk.
template<bool RELU, bool OUT_BF16>
__global__ void gemm_bf16_kernel(const bf16* __restrict__ A, const bf16* __restrict__ Bp,
                                 void* __restrict__ Cout, int M, int Kdim, int Ndim) {
  const int lane  = threadIdx.x & 31;
  const int mTile = blockIdx.x * (blockDim.x >> 5) + (threadIdx.x >> 5);
  if (mTile * 16 >= M) return;                 // uniform per wave: EXEC stays full
  const int mBase  = mTile * 16;
  const int nTiles = Ndim >> 4;
  const int kTiles = Kdim >> 5;
  for (int nc = 0; nc < nTiles; nc += 8) {
    v8f acc[8];
    #pragma unroll
    for (int nt = 0; nt < 8; ++nt)
      #pragma unroll
      for (int e = 0; e < 8; ++e) acc[nt][e] = 0.0f;
    for (int kt = 0; kt < kTiles; ++kt) {
      const v16bf a = load_a16x32(A, Kdim, mBase, kt << 5, lane);
      #pragma unroll
      for (int nt = 0; nt < 8; ++nt) {
        const v16bf b = *(const v16bf*)(Bp +
            (((size_t)kt * nTiles + (size_t)(nc + nt)) * 32 + lane) * 16);
        acc[nt] = wmma_bf16(a, b, acc[nt]);
      }
    }
    // D layout: VGPR j -> row mBase+j (lanes 0-15) / mBase+8+j (lanes 16-31), col = lane&15
    const int colLane = lane & 15;
    const int rowBase = mBase + ((lane >> 4) << 3);
    #pragma unroll
    for (int nt = 0; nt < 8; ++nt) {
      const int col = (nc + nt) * 16 + colLane;
      #pragma unroll
      for (int j = 0; j < 8; ++j) {
        float val = acc[nt][j];
        if (RELU) val = fmaxf(val, 0.0f);
        const size_t idx = (size_t)(rowBase + j) * Ndim + col;
        if (OUT_BF16) ((bf16*)Cout)[idx] = (bf16)val;
        else          ((float*)Cout)[idx] = val;
      }
    }
  }
}

// ------------------------------------------------- tiny precompute kernels
// 6-row, 3-layer angle MLP: exact factorization of the T-side 39 GFLOP MLP.
__global__ void table6_kernel(const float* __restrict__ ang_table,
                              const float* __restrict__ W0, const float* __restrict__ b0,
                              const float* __restrict__ W1, const float* __restrict__ b1,
                              const float* __restrict__ W2, const float* __restrict__ b2,
                              float* __restrict__ out) {
  __shared__ float buf0[6][128];
  __shared__ float buf1[6][128];
  const int c = threadIdx.x;  // 128 threads
  for (int r = 0; r < 6; ++r) buf0[r][c] = ang_table[r * 128 + c];
  __syncthreads();
  const float* Ws[3] = {W0, W1, W2};
  const float* bs[3] = {b0, b1, b2};
  for (int layer = 0; layer < 3; ++layer) {
    float (*src)[128] = (layer & 1) ? buf1 : buf0;
    float (*dst)[128] = (layer & 1) ? buf0 : buf1;
    const float* W = Ws[layer];
    const float* bb = bs[layer];
    for (int r = 0; r < 6; ++r) {
      float s = bb[c];
      for (int k = 0; k < 128; ++k) s += src[r][k] * W[(size_t)k * 128 + c];
      dst[r][c] = fmaxf(s, 0.0f);
    }
    __syncthreads();
  }
  for (int r = 0; r < 6; ++r) out[r * 128 + c] = buf1[r][c];  // 3 layers end in buf1
}

// Repack fp32 row-major W[K x N] into the WMMA B-matrix lane layout (bf16).
__global__ void packB_kernel(const float* __restrict__ W, bf16* __restrict__ out,
                             int Kdim, int Ndim) {
  const int tidx = blockIdx.x * blockDim.x + threadIdx.x;
  if (tidx >= Kdim * Ndim) return;
  const int i      = tidx & 15;
  const int lane   = (tidx >> 4) & 31;
  const int tile   = tidx >> 9;
  const int nTiles = Ndim >> 4;
  const int nTile  = tile % nTiles;
  const int kTile  = tile / nTiles;
  const int n = nTile * 16 + (lane & 15);
  const int k = kTile * 32 + ((lane >> 4) << 3) + ((i < 8) ? i : (i - 8 + 16));
  out[tidx] = (bf16)W[(size_t)k * Ndim + n];
}

__global__ void cvt_bf16_kernel(const float* __restrict__ in, bf16* __restrict__ out,
                                size_t n) {
  size_t i = (size_t)blockIdx.x * blockDim.x + threadIdx.x;
  if (i < n) out[i] = (bf16)in[i];
}

// ------------------------------------------------------------- triplet side
__global__ void angle_bin_kernel(const float* __restrict__ pos,
                                 const int* __restrict__ idx_i,
                                 const int* __restrict__ idx_j,
                                 const int* __restrict__ idx_k,
                                 int* __restrict__ bins, int Tn) {
  const int t = blockIdx.x * blockDim.x + threadIdx.x;
  if (t >= Tn) return;
  const int ii = idx_i[t], jj = idx_j[t], kk = idx_k[t];
  const float ix = pos[3 * ii], iy = pos[3 * ii + 1], iz = pos[3 * ii + 2];
  const float jx = pos[3 * jj] - ix, jy = pos[3 * jj + 1] - iy, jz = pos[3 * jj + 2] - iz;
  const float kx = pos[3 * kk] - ix, ky = pos[3 * kk + 1] - iy, kz = pos[3 * kk + 2] - iz;
  const float a = jx * kx + jy * ky + jz * kz;
  const float cx = jy * kz - jz * ky;
  const float cy = jz * kx - jx * kz;
  const float cz = jx * ky - jy * kx;
  const float b = sqrtf(cx * cx + cy * cy + cz * cz);
  int bin = (int)(atan2f(b, a) / (PI_F / 6.0f));
  bins[t] = min(max(bin, 0), 5);
}

// Pass 1: per (triplet, head) exp-attention + denominator scatter-add.
__global__ void attn1_kernel(const float* __restrict__ K, const float* __restrict__ Q,
                             const float* __restrict__ tab, const int* __restrict__ bins,
                             const int* __restrict__ index_kj, const int* __restrict__ index_ji,
                             const float* __restrict__ basic_attn, const float* __restrict__ dis_W,
                             float* __restrict__ att, float* __restrict__ denom, int Tn) {
  const int g = blockIdx.x * blockDim.x + threadIdx.x;
  if (g >= Tn * 8) return;
  const int t = g >> 3, h = g & 7;
  const int kj = index_kj[t], ji = index_ji[t], bin = bins[t];
  const float4* k4 = (const float4*)(K + (size_t)kj * 128 + h * 16);
  const float4* q4 = (const float4*)(Q + (size_t)ji * 128 + h * 16);
  const float4* a4 = (const float4*)(tab + bin * 128 + h * 16);
  float dot = 0.0f;
  #pragma unroll
  for (int i = 0; i < 4; ++i) {
    const float4 kv = k4[i], qv = q4[i], av = a4[i];
    dot += (kv.x + av.x) * (qv.x + av.x);
    dot += (kv.y + av.y) * (qv.y + av.y);
    dot += (kv.z + av.z) * (qv.z + av.z);
    dot += (kv.w + av.w) * (qv.w + av.w);
  }
  const float e = expf(dot * 0.25f + basic_attn[kj] * dis_W[h]);  // 1/sqrt(F)=0.25
  att[g] = e;
  atomicAdd(&denom[(size_t)ji * 8 + h], e);
}

// Pass 2: normalized weighted value scatter-add into msg[E,128].
__global__ void attn2_kernel(const float* __restrict__ V, const float* __restrict__ tab,
                             const int* __restrict__ bins, const int* __restrict__ index_kj,
                             const int* __restrict__ index_ji, const float* __restrict__ att,
                             const float* __restrict__ denom, float* __restrict__ msg, int Tn) {
  const int g = blockIdx.x * blockDim.x + threadIdx.x;
  if (g >= Tn * 8) return;
  const int t = g >> 3, h = g & 7;
  const int kj = index_kj[t], ji = index_ji[t], bin = bins[t];
  const float w = att[g] / denom[(size_t)ji * 8 + h];
  const float4* v4 = (const float4*)(V + (size_t)kj * 128 + h * 16);
  const float4* a4 = (const float4*)(tab + bin * 128 + h * 16);
  float* m = msg + (size_t)ji * 128 + h * 16;
  #pragma unroll
  for (int i = 0; i < 4; ++i) {
    const float4 vv = v4[i], av = a4[i];
    atomicAdd(&m[i * 4 + 0], (vv.x + av.x) * w);
    atomicAdd(&m[i * 4 + 1], (vv.y + av.y) * w);
    atomicAdd(&m[i * 4 + 2], (vv.z + av.z) * w);
    atomicAdd(&m[i * 4 + 3], (vv.w + av.w) * w);
  }
}

// ----------------------------------------------------------- layernorm side
// msg += bond (-> he, kept in place), LN1(he) -> bf16 for the FFN GEMM.
__global__ void he_ln1_kernel(float* __restrict__ msg, const float* __restrict__ bond,
                              const float* __restrict__ g1, const float* __restrict__ b1,
                              bf16* __restrict__ hln, int E) {
  const int wave = blockIdx.x * (blockDim.x >> 5) + (threadIdx.x >> 5);
  const int lane = threadIdx.x & 31;
  if (wave >= E) return;
  const size_t base = (size_t)wave * 128 + lane * 4;
  float x[4], s = 0.0f;
  #pragma unroll
  for (int i = 0; i < 4; ++i) {
    x[i] = msg[base + i] + bond[base + i];
    msg[base + i] = x[i];
    s += x[i];
  }
  for (int o = 16; o >= 1; o >>= 1) s += __shfl_xor(s, o);
  const float mean = s * (1.0f / 128.0f);
  float vs = 0.0f;
  #pragma unroll
  for (int i = 0; i < 4; ++i) { const float d = x[i] - mean; vs += d * d; }
  for (int o = 16; o >= 1; o >>= 1) vs += __shfl_xor(vs, o);
  const float rstd = rsqrtf(vs * (1.0f / 128.0f) + 1e-5f);
  const int c = lane * 4;
  #pragma unroll
  for (int i = 0; i < 4; ++i)
    hln[base + i] = (bf16)((x[i] - mean) * rstd * g1[c + i] + b1[c + i]);
}

// out = LN2(y + he)
__global__ void ln2_kernel(const float* __restrict__ y, const float* __restrict__ he,
                           const float* __restrict__ g2, const float* __restrict__ b2,
                           float* __restrict__ out, int E) {
  const int wave = blockIdx.x * (blockDim.x >> 5) + (threadIdx.x >> 5);
  const int lane = threadIdx.x & 31;
  if (wave >= E) return;
  const size_t base = (size_t)wave * 128 + lane * 4;
  float x[4], s = 0.0f;
  #pragma unroll
  for (int i = 0; i < 4; ++i) { x[i] = y[base + i] + he[base + i]; s += x[i]; }
  for (int o = 16; o >= 1; o >>= 1) s += __shfl_xor(s, o);
  const float mean = s * (1.0f / 128.0f);
  float vs = 0.0f;
  #pragma unroll
  for (int i = 0; i < 4; ++i) { const float d = x[i] - mean; vs += d * d; }
  for (int o = 16; o >= 1; o >>= 1) vs += __shfl_xor(vs, o);
  const float rstd = rsqrtf(vs * (1.0f / 128.0f) + 1e-5f);
  const int c = lane * 4;
  #pragma unroll
  for (int i = 0; i < 4; ++i)
    out[base + i] = (x[i] - mean) * rstd * g2[c + i] + b2[c + i];
}

// ------------------------------------------------------------------ launch
extern "C" void kernel_launch(void* const* d_in, const int* in_sizes, int n_in,
                              void* d_out, int out_size, void* d_ws, size_t ws_size,
                              hipStream_t stream) {
  const float* bond    = (const float*)d_in[0];
  const float* pos     = (const float*)d_in[1];
  const float* battn   = (const float*)d_in[2];
  const float* angtab  = (const float*)d_in[3];
  const float* ang_inW = (const float*)d_in[4];
  const float* ang_inb = (const float*)d_in[5];
  const float* ang2W   = (const float*)d_in[6];
  const float* ang2b   = (const float*)d_in[7];
  const float* ang1W   = (const float*)d_in[8];
  const float* ang1b   = (const float*)d_in[9];
  const float* kW      = (const float*)d_in[10];
  const float* qW      = (const float*)d_in[11];
  const float* vW      = (const float*)d_in[12];
  const float* disW    = (const float*)d_in[13];
  const float* W1      = (const float*)d_in[14];
  const float* W2      = (const float*)d_in[15];
  const float* ln1g    = (const float*)d_in[16];
  const float* ln1b    = (const float*)d_in[17];
  const float* ln2g    = (const float*)d_in[18];
  const float* ln2b    = (const float*)d_in[19];
  const int* index_kj  = (const int*)d_in[20];
  const int* index_ji  = (const int*)d_in[21];
  const int* idx_i     = (const int*)d_in[22];
  const int* idx_j     = (const int*)d_in[23];
  const int* idx_k     = (const int*)d_in[24];

  const int E = in_sizes[0] / 128;   // 200000
  const int T = in_sizes[20];        // 400000

  auto al = [](size_t x) { return (x + 255) & ~(size_t)255; };
  char* ws = (char*)d_ws;
  const size_t szED = al((size_t)E * 128 * sizeof(float));
  size_t off = 0;
  const size_t off_K   = off; off += szED;
  const size_t off_Q   = off; off += szED;
  const size_t off_V   = off; off += szED;
  const size_t off_he  = off; off += szED;                               // msg -> he
  const size_t off_att = off; off += al((size_t)T * 8 * sizeof(float));
  const size_t off_den = off; off += al((size_t)E * 8 * sizeof(float));
  const size_t off_bin = off; off += al((size_t)T * sizeof(int));
  const size_t off_bbf = off; off += al((size_t)E * 128 * sizeof(bf16)); // bond bf16
  const size_t off_hln = off; off += al((size_t)E * 128 * sizeof(bf16));
  const size_t off_tab = off; off += al(6 * 128 * sizeof(float));
  const size_t off_kWp = off; off += al(128 * 128 * sizeof(bf16));
  const size_t off_qWp = off; off += al(128 * 128 * sizeof(bf16));
  const size_t off_vWp = off; off += al(128 * 128 * sizeof(bf16));
  const size_t off_W1p = off; off += al(128 * 256 * sizeof(bf16));
  const size_t off_W2p = off; off += al(256 * 128 * sizeof(bf16));
  const size_t off_hid = off_K;   // E*256 bf16 == E*128 f32 bytes; K dead after attn1
  const size_t off_y   = off_Q;   // Q dead after attn1
  (void)ws_size; (void)n_in; (void)out_size; (void)idx_j;

  float* Kb   = (float*)(ws + off_K);
  float* Qb   = (float*)(ws + off_Q);
  float* Vb   = (float*)(ws + off_V);
  float* msg  = (float*)(ws + off_he);
  float* att  = (float*)(ws + off_att);
  float* den  = (float*)(ws + off_den);
  int*   bins = (int*)  (ws + off_bin);
  bf16*  bbf  = (bf16*) (ws + off_bbf);
  bf16*  hln  = (bf16*) (ws + off_hln);
  float* tab  = (float*)(ws + off_tab);
  bf16*  kWp  = (bf16*) (ws + off_kWp);
  bf16*  qWp  = (bf16*) (ws + off_qWp);
  bf16*  vWp  = (bf16*) (ws + off_vWp);
  bf16*  W1p  = (bf16*) (ws + off_W1p);
  bf16*  W2p  = (bf16*) (ws + off_W2p);
  bf16*  hid  = (bf16*) (ws + off_hid);
  float* yb   = (float*)(ws + off_y);
  float* out  = (float*)d_out;

  // 1. angle table (exact collapse of the T-side MLP)
  table6_kernel<<<1, 128, 0, stream>>>(angtab, ang_inW, ang_inb, ang2W, ang2b,
                                       ang1W, ang1b, tab);
  // 2. weight repack + bond bf16 convert
  packB_kernel<<<(128 * 128 + 255) / 256, 256, 0, stream>>>(kW, kWp, 128, 128);
  packB_kernel<<<(128 * 128 + 255) / 256, 256, 0, stream>>>(qW, qWp, 128, 128);
  packB_kernel<<<(128 * 128 + 255) / 256, 256, 0, stream>>>(vW, vWp, 128, 128);
  packB_kernel<<<(128 * 256 + 255) / 256, 256, 0, stream>>>(W1, W1p, 128, 256);
  packB_kernel<<<(256 * 128 + 255) / 256, 256, 0, stream>>>(W2, W2p, 256, 128);
  {
    const size_t n = (size_t)E * 128;
    cvt_bf16_kernel<<<(unsigned)((n + 255) / 256), 256, 0, stream>>>(bond, bbf, n);
  }
  // 3. K/Q/V WMMA GEMMs
  const int mWaves  = (E + 15) / 16;
  const int gemmBlk = (mWaves + 7) / 8;
  gemm_bf16_kernel<false, false><<<gemmBlk, 256, 0, stream>>>(bbf, kWp, Kb, E, 128, 128);
  gemm_bf16_kernel<false, false><<<gemmBlk, 256, 0, stream>>>(bbf, qWp, Qb, E, 128, 128);
  gemm_bf16_kernel<false, false><<<gemmBlk, 256, 0, stream>>>(bbf, vWp, Vb, E, 128, 128);
  // 4. triplet side
  angle_bin_kernel<<<(T + 255) / 256, 256, 0, stream>>>(pos, idx_i, idx_j, idx_k, bins, T);
  hipMemsetAsync(den, 0, (size_t)E * 8 * sizeof(float), stream);
  hipMemsetAsync(msg, 0, (size_t)E * 128 * sizeof(float), stream);
  attn1_kernel<<<(T * 8 + 255) / 256, 256, 0, stream>>>(Kb, Qb, tab, bins, index_kj,
                                                        index_ji, battn, disW, att, den, T);
  attn2_kernel<<<(T * 8 + 255) / 256, 256, 0, stream>>>(Vb, tab, bins, index_kj,
                                                        index_ji, att, den, msg, T);
  // 5. he = msg + bond, LN1 -> bf16
  he_ln1_kernel<<<(E + 7) / 8, 256, 0, stream>>>(msg, bond, ln1g, ln1b, hln, E);
  // 6. FFN WMMA GEMMs + final LN
  gemm_bf16_kernel<true,  true ><<<gemmBlk, 256, 0, stream>>>(hln, W1p, hid, E, 128, 256);
  gemm_bf16_kernel<false, false><<<gemmBlk, 256, 0, stream>>>(hid, W2p, yb,  E, 256, 128);
  ln2_kernel<<<(E + 7) / 8, 256, 0, stream>>>(yb, msg, ln2g, ln2b, out, E);
}
